// CurveNet_73315091743161
// MI455X (gfx1250) — matrix operand order repaired and checked by reading, not compiled, not run on hardware
//
#include <hip/hip_runtime.h>

#define KNN 20

typedef __attribute__((ext_vector_type(16))) _Float16 v16h;
typedef __attribute__((ext_vector_type(8)))  float    v8f;
typedef __attribute__((ext_vector_type(4)))  float    v4f;

// =====================================================================
// Generic f16-WMMA GEMM with fused epilogue.
//   D[M,N] = epi( A[M,K] @ W[K,N] )
// epi: optional per-col scale g / bias b (eval batchnorm), optional residual
// R (direct row, or indirect row via Rrow gather -> fuses gather_nd(x, idx)),
// activation (0 none / 1 relu / 2 leaky-0.2), optional mean over rowDiv
// consecutive rows via atomicAdd (D pre-zeroed by caller).
//
// NT = N-tiles per wave. NT=2 reuses each A fragment across two back-to-back
// v_wmma ops (two 16x16 accumulators), halving A-side traffic per MAC; the
// host selects NT=2 whenever N%32==0 so those launches only see interior
// N-tiles. Tile coords are forced into SGPRs via readfirstlane so fast/slow
// path selection is a scalar branch: interior tiles run only b128 vector
// loads + v_cvt + v_wmma per K-step; edge tiles use clamped-index loads
// (v_min + value select) -- no exec-mask branching in the hot loop.
// =====================================================================
template <int NT>
__global__ __launch_bounds__(128)
void gemm_wmma_kernel(const float* __restrict__ A, const float* __restrict__ W,
                      float* __restrict__ D, int M, int N, int Kd,
                      const float* __restrict__ g, const float* __restrict__ bia,
                      const float* __restrict__ R, const int* __restrict__ Rrow,
                      int act, int rowDiv)
{
  const int lane = threadIdx.x & 31;
  const int wave = threadIdx.x >> 5;
  // Scalarize tile coordinates (uniform per wave, unprovable from threadIdx).
  const int tM = __builtin_amdgcn_readfirstlane(blockIdx.y * 4 + wave);
  const int tN = __builtin_amdgcn_readfirstlane(blockIdx.x);
  const int row0 = tM * 16, col0 = tN * (16 * NT);
  if (row0 >= M) return;                 // scalar branch

  const int arow  = row0 + (lane & 15);  // A row serviced by this lane
  const int khalf = (lane >> 4) * 8;     // K sub-block per lane half
  const bool kAligned = ((Kd & 3) == 0);
  const bool mFull = (row0 + 16 <= M);

  const bool rowOk = (arow < M);
  const float* __restrict__ arowBase = A + (size_t)(rowOk ? arow : 0) * Kd;
  const int kClamp = Kd - 1;

  v8f acc[NT];
#pragma unroll
  for (int t = 0; t < NT; ++t) acc[t] = (v8f){0.f,0.f,0.f,0.f,0.f,0.f,0.f,0.f};

  for (int k0 = 0; k0 < Kd; k0 += 32) {
    const bool kFull = (k0 + 32 <= Kd);
    // ---- A fragment: lane holds K runs [khalf,khalf+8) and [16+khalf,+8) ---
    v16h a;
    if (mFull && kFull && kAligned) {          // scalar branch
      const float* pa = arowBase + k0 + khalf;
      v4f x0 = *(const v4f*)(pa);
      v4f x1 = *(const v4f*)(pa + 4);
      v4f x2 = *(const v4f*)(pa + 16);
      v4f x3 = *(const v4f*)(pa + 20);
      __builtin_prefetch(pa + 32, 0, 1);       // speculative: drop-safe
#pragma unroll
      for (int j = 0; j < 4; ++j) {
        a[j]      = (_Float16)x0[j];
        a[4 + j]  = (_Float16)x1[j];
        a[8 + j]  = (_Float16)x2[j];
        a[12 + j] = (_Float16)x3[j];
      }
    } else {
#pragma unroll
      for (int e = 0; e < 16; ++e) {
        int kk = ((e < 8) ? e : (8 + e)) + khalf;  // khalf+e | 16+khalf+(e-8)
        int kg = k0 + kk;
        bool ok = rowOk && (kg < Kd);
        float v = arowBase[kg < kClamp ? kg : kClamp];  // clamped, always safe
        a[e] = (_Float16)(ok ? v : 0.0f);
      }
    }
    // ---- B fragments: lane = N col (lane&15), K runs same as A fragment ---
#pragma unroll
    for (int t = 0; t < NT; ++t) {
      const int colT = col0 + 16 * t;
      const int cg = colT + (lane & 15);
      v16h b;
      if ((colT + 16 <= N) && kFull) {          // scalar branch
#pragma unroll
        for (int e = 0; e < 16; ++e) {
          int kk = ((e < 8) ? e : (8 + e)) + khalf;
          b[e] = (_Float16)W[(size_t)(k0 + kk) * N + cg];
        }
      } else {
#pragma unroll
        for (int e = 0; e < 16; ++e) {
          int kk = ((e < 8) ? e : (8 + e)) + khalf;
          int kg = k0 + kk;
          bool ok = (kg < Kd) && (cg < N);
          float v = W[(size_t)(kg < kClamp ? kg : kClamp) * N + (cg < N - 1 ? cg : N - 1)];  // clamped, always safe
          b[e] = (_Float16)(ok ? v : 0.0f);
        }
      }
      acc[t] = __builtin_amdgcn_wmma_f32_16x16x32_f16(false, a, false, b,
                                                      (short)0, acc[t],
                                                      false, false);
    }
  }

  // ---- epilogue: C/D layout -> lane = N col, vgpr r = M row (+8 per half) --
  const int rbase = row0 + ((lane >> 4) << 3);
#pragma unroll
  for (int t = 0; t < NT; ++t) {
    const int ccol = col0 + 16 * t + (lane & 15);
    if (ccol >= N) continue;
#pragma unroll
    for (int r = 0; r < 8; ++r) {
      int rr = rbase + r;
      if (rr >= M) continue;
      float v = acc[t][r];
      if (g)   v *= g[ccol];
      if (bia) v += bia[ccol];
      if (R) {
        size_t sr = Rrow ? (size_t)Rrow[rr] : (size_t)rr;
        v += R[sr * (size_t)N + ccol];
      }
      if (act == 1)      v = fmaxf(v, 0.0f);
      else if (act == 2) v = (v > 0.0f) ? v : 0.2f * v;
      if (rowDiv > 1) atomicAdd(&D[(size_t)(rr / rowDiv) * N + ccol], v / (float)rowDiv);
      else            D[(size_t)rr * N + ccol] = v;
    }
  }
}

// =====================================================================
// kNN: per (b, query) insertion top-20 of squared distance.
// =====================================================================
__global__ void knn_kernel(const float* __restrict__ q, const float* __restrict__ r,
                           int Nq, int Nr, int* __restrict__ outIdx)
{
  int t = blockIdx.x * blockDim.x + threadIdx.x;
  if (t >= 16 * Nq) return;
  int b = t / Nq, qi = t % Nq;
  const float* qp = q + ((size_t)b * Nq + qi) * 3;
  float qx = qp[0], qy = qp[1], qz = qp[2];
  float bd[KNN]; int bi[KNN];
  for (int j = 0; j < KNN; ++j) { bd[j] = 3.4e38f; bi[j] = 0; }
  const float* rb = r + (size_t)b * Nr * 3;
  for (int j = 0; j < Nr; ++j) {
    float dx = qx - rb[j * 3 + 0];
    float dy = qy - rb[j * 3 + 1];
    float dz = qz - rb[j * 3 + 2];
    float d = dx * dx + dy * dy + dz * dz;
    if (d < bd[KNN - 1]) {
      int p = KNN - 1;
      while (p > 0 && bd[p - 1] > d) { bd[p] = bd[p - 1]; bi[p] = bi[p - 1]; --p; }
      bd[p] = d; bi[p] = j;
    }
  }
  int* o = outIdx + (size_t)t * KNN;
  for (int j = 0; j < KNN; ++j) o[j] = bi[j];
}

// =====================================================================
// Farthest point sampling: one block per batch, dist array in LDS.
// =====================================================================
__global__ void fps_kernel(const float* __restrict__ xyz, int N, int npoint,
                           int* __restrict__ sidx)
{
  __shared__ float dist[2048];
  __shared__ float rv[256];
  __shared__ int   ri[256];
  __shared__ int   s_last;
  int b = blockIdx.x, tid = threadIdx.x;
  for (int i = tid; i < N; i += 256) dist[i] = 1e10f;
  if (tid == 0) { s_last = 0; sidx[b * npoint] = 0; }
  __syncthreads();
  const float* xb = xyz + (size_t)b * N * 3;
  for (int it = 1; it < npoint; ++it) {
    int last = s_last;
    float lx = xb[last * 3], ly = xb[last * 3 + 1], lz = xb[last * 3 + 2];
    float bv = -1.0f; int bj = 0;
    for (int i = tid; i < N; i += 256) {
      float dx = xb[i * 3] - lx, dy = xb[i * 3 + 1] - ly, dz = xb[i * 3 + 2] - lz;
      float d = fminf(dist[i], dx * dx + dy * dy + dz * dz);
      dist[i] = d;
      if (d > bv) { bv = d; bj = i; }
    }
    rv[tid] = bv; ri[tid] = bj;
    __syncthreads();
    for (int s = 128; s > 0; s >>= 1) {
      if (tid < s) {
        if (rv[tid + s] > rv[tid] ||
            (rv[tid + s] == rv[tid] && ri[tid + s] < ri[tid])) {
          rv[tid] = rv[tid + s]; ri[tid] = ri[tid + s];
        }
      }
      __syncthreads();
    }
    if (tid == 0) { s_last = ri[0]; sidx[b * npoint + it] = ri[0]; }
    __syncthreads();
  }
}

__global__ void build_srow_kernel(const int* __restrict__ sidx, int np, int N,
                                  int* __restrict__ srow)
{
  int t = blockIdx.x * blockDim.x + threadIdx.x;
  if (t >= 16 * np) return;
  srow[t] = (t / np) * N + sidx[t];
}

__global__ void gather_rows_kernel(const float* __restrict__ src, const int* __restrict__ rows,
                                   int n, int C, float* __restrict__ dst)
{
  int t = blockIdx.x * blockDim.x + threadIdx.x;
  if (t >= n * C) return;
  int r = t / C, c = t % C;
  dst[t] = src[(size_t)rows[r] * C + c];
}

// =====================================================================
// Radius-masked max pool over kNN neighborhood (k==0 always valid).
// =====================================================================
__global__ void masked_pool_kernel(const float* __restrict__ nxyz, const float* __restrict__ xyz,
                                   const int* __restrict__ nidx, const float* __restrict__ x,
                                   int Np, int N, int C, float r2, float* __restrict__ out)
{
  int t = blockIdx.x * blockDim.x + threadIdx.x;
  if (t >= 16 * Np * C) return;
  int c = t % C; int i = (t / C) % Np; int b = t / (C * Np);
  const float* nc = nxyz + ((size_t)b * Np + i) * 3;
  float cx = nc[0], cy = nc[1], cz = nc[2];
  const int* id = nidx + ((size_t)b * Np + i) * KNN;
  float best = -1e9f;
  for (int k = 0; k < KNN; ++k) {
    int ni = id[k];
    const float* p = xyz + ((size_t)b * N + ni) * 3;
    float dx = p[0] - cx, dy = p[1] - cy, dz = p[2] - cz;
    bool valid = (k == 0) || (dx * dx + dy * dy + dz * dz <= r2);
    float v = valid ? x[((size_t)b * N + ni) * C + c] : -1e9f;
    best = fmaxf(best, v);
  }
  out[t] = best;
}

// =====================================================================
// Fused lpfa_initial: relpos(9) @ W(9x32) + BN + lrelu, max over k.
// =====================================================================
__global__ void lpfa0_kernel(const float* __restrict__ xyz, const int* __restrict__ idx,
                             const float* __restrict__ W, const float* __restrict__ g,
                             const float* __restrict__ bb, int N, float* __restrict__ out)
{
  int t = blockIdx.x * blockDim.x + threadIdx.x;
  if (t >= 16 * N * 32) return;
  int c = t & 31; int n = (t >> 5) % N; int b = t / (32 * N);
  const float* ctr = xyz + ((size_t)b * N + n) * 3;
  float cx = ctr[0], cy = ctr[1], cz = ctr[2];
  const int* id = idx + ((size_t)b * N + n) * KNN;
  float best = -3.4e38f;
  for (int k = 0; k < KNN; ++k) {
    const float* p = xyz + ((size_t)b * N + id[k]) * 3;
    float px = p[0], py = p[1], pz = p[2];
    float f9[9] = {cx, cy, cz, px, py, pz, px - cx, py - cy, pz - cz};
    float s = 0.f;
#pragma unroll
    for (int j = 0; j < 9; ++j) s += f9[j] * W[j * 32 + c];
    s = s * g[c] + bb[c];
    s = (s > 0.f) ? s : 0.2f * s;
    best = fmaxf(best, s);
  }
  out[t] = best;
}

// rel-pos 9-feature rows + flat neighbor-row map (for gathered residual).
__global__ void relpos_kernel(const float* __restrict__ xyz, const int* __restrict__ idx,
                              int N, float* __restrict__ f9, int* __restrict__ nbrrow)
{
  int t = blockIdx.x * blockDim.x + threadIdx.x;
  if (t >= 16 * N * KNN) return;
  int rem = t / KNN; int n = rem % N; int b = rem / N;
  int ni = idx[t];
  const float* c = xyz + ((size_t)b * N + n) * 3;
  const float* p = xyz + ((size_t)b * N + ni) * 3;
  float* o = f9 + (size_t)t * 9;
  o[0] = c[0]; o[1] = c[1]; o[2] = c[2];
  o[3] = p[0]; o[4] = p[1]; o[5] = p[2];
  o[6] = p[0] - c[0]; o[7] = p[1] - c[1]; o[8] = p[2] - c[2];
  nbrrow[t] = b * N + ni;
}

__global__ void zero_kernel(float* __restrict__ p, int n)
{
  int t = blockIdx.x * blockDim.x + threadIdx.x;
  if (t < n) p[t] = 0.f;
}

// ============================= curve ops ==============================
__global__ void score_kernel(const float* __restrict__ h, const float* __restrict__ Ws,
                             int total, int C, float* __restrict__ sco)
{
  int t = blockIdx.x * blockDim.x + threadIdx.x;
  if (t >= total) return;
  const float* x = h + (size_t)t * C;
  float s = 0.f;
  for (int c = 0; c < C; ++c) s += x[c] * Ws[c];
  sco[t] = s;
}

__global__ void topk_starts_kernel(const float* __restrict__ score, int N, int M,
                                   int* __restrict__ starts)
{
  __shared__ float s[2048];
  __shared__ float rv[256];
  __shared__ int   ri[256];
  int b = blockIdx.x, tid = threadIdx.x;
  for (int i = tid; i < N; i += 256) s[i] = score[(size_t)b * N + i];
  __syncthreads();
  for (int m = 0; m < M; ++m) {
    float bv = -3.4e38f; int bj = 0;
    for (int i = tid; i < N; i += 256)
      if (s[i] > bv) { bv = s[i]; bj = i; }
    rv[tid] = bv; ri[tid] = bj;
    __syncthreads();
    for (int st = 128; st > 0; st >>= 1) {
      if (tid < st) {
        if (rv[tid + st] > rv[tid] ||
            (rv[tid + st] == rv[tid] && ri[tid + st] < ri[tid])) {
          rv[tid] = rv[tid + st]; ri[tid] = ri[tid + st];
        }
      }
      __syncthreads();
    }
    if (tid == 0) { starts[b * M + m] = ri[0]; s[ri[0]] = -3.4e38f; }
    __syncthreads();
  }
}

// Momentum curve walk; directly accumulates mean over the L steps -> cmean.
__global__ void curve_walk_kernel(const float* __restrict__ h, const int* __restrict__ idx,
                                  const int* __restrict__ starts, int N, int C, int M, int L,
                                  float* __restrict__ cmean)
{
  int t = blockIdx.x * blockDim.x + threadIdx.x;
  if (t >= 16 * M) return;
  int b = t / M;
  int cur = starts[t];
  const float* hb = h + (size_t)b * N * C;
  float st[32], acc[32];
  for (int c = 0; c < C; ++c) { st[c] = hb[(size_t)cur * C + c]; acc[c] = 0.f; }
  for (int l = 0; l < L; ++l) {
    float best = -3.4e38f; int bestni = cur;
    const int* id = idx + ((size_t)b * N + cur) * KNN;
    for (int k = 0; k < KNN; ++k) {
      int ni = id[k];
      const float* nf = hb + (size_t)ni * C;
      float d = 0.f;
      for (int c = 0; c < C; ++c) d += st[c] * nf[c];
      if (d > best) { best = d; bestni = ni; }
    }
    const float* nf = hb + (size_t)bestni * C;
    for (int c = 0; c < C; ++c) {
      float v = nf[c];
      acc[c] += v;
      st[c] = 0.5f * st[c] + 0.5f * v;
    }
    cur = bestni;
  }
  float inv = 1.0f / (float)L;
  float* o = cmean + (size_t)t * C;
  for (int c = 0; c < C; ++c) o[c] = acc[c] * inv;
}

// softmax attention over curve means; writes agg (Wa matmul + residual fused
// later in the WMMA GEMM epilogue).
__global__ void curve_att_kernel(const float* __restrict__ h, const float* __restrict__ cm,
                                 int N, int C, int M, float* __restrict__ agg)
{
  int t = blockIdx.x * blockDim.x + threadIdx.x;
  if (t >= 16 * N) return;
  int b = t / N;
  const float* xi = h + (size_t)t * C;
  float x[32];
  for (int c = 0; c < C; ++c) x[c] = xi[c];
  float scale = rsqrtf((float)C);
  const float* cb = cm + (size_t)b * M * C;
  float mx = -3.4e38f;
  for (int m = 0; m < M; ++m) {
    const float* cr = cb + (size_t)m * C;
    float d = 0.f;
    for (int c = 0; c < C; ++c) d += x[c] * cr[c];
    mx = fmaxf(mx, d * scale);
  }
  float a[32];
  for (int c = 0; c < C; ++c) a[c] = 0.f;
  float sum = 0.f;
  for (int m = 0; m < M; ++m) {
    const float* cr = cb + (size_t)m * C;
    float d = 0.f;
    for (int c = 0; c < C; ++c) d += x[c] * cr[c];
    float w = __expf(d * scale - mx);
    sum += w;
    for (int c = 0; c < C; ++c) a[c] += w * cr[c];
  }
  float inv = 1.0f / sum;
  float* o = agg + (size_t)t * C;
  for (int c = 0; c < C; ++c) o[c] = a[c] * inv;
}

// final global max+mean pool over the 64 remaining points -> f [16,2048]
__global__ void final_pool_kernel(const float* __restrict__ h, int C, float* __restrict__ f)
{
  int t = blockIdx.x * blockDim.x + threadIdx.x;
  if (t >= 16 * C) return;
  int b = t / C, c = t % C;
  float mx = -3.4e38f, sm = 0.f;
  for (int i = 0; i < 64; ++i) {
    float v = h[((size_t)b * 64 + i) * C + c];
    mx = fmaxf(mx, v); sm += v;
  }
  f[(size_t)b * 2 * C + c] = mx;
  f[(size_t)b * 2 * C + C + c] = sm * (1.0f / 64.0f);
}

// =====================================================================
// Host orchestration
// =====================================================================
static inline dim3 g1d(int n) { return dim3((unsigned)((n + 255) / 256)); }

static void gemmL(hipStream_t s, const float* A, const float* W, float* D,
                  int M, int N, int K, const float* g, const float* b,
                  const float* R, const int* Rrow, int act, int rowDiv)
{
  if ((N & 31) == 0) {   // two N-tiles per wave: A-fragment reuse across WMMAs
    dim3 grid((unsigned)(N / 32), (unsigned)((M + 63) / 64));
    gemm_wmma_kernel<2><<<grid, 128, 0, s>>>(A, W, D, M, N, K, g, b, R, Rrow,
                                             act, rowDiv);
  } else {
    dim3 grid((unsigned)((N + 15) / 16), (unsigned)((M + 63) / 64));
    gemm_wmma_kernel<1><<<grid, 128, 0, s>>>(A, W, D, M, N, K, g, b, R, Rrow,
                                             act, rowDiv);
  }
}

extern "C" void kernel_launch(void* const* d_in, const int* in_sizes, int n_in,
                              void* d_out, int out_size, void* d_ws, size_t ws_size,
                              hipStream_t stream)
{
  (void)in_sizes; (void)n_in; (void)out_size; (void)ws_size;

  // ---- unpack params (JAX pytree sorted-key order after xyz) ----
  int pi = 0;
  auto nx = [&]() { return (const float*)d_in[pi++]; };
  const float* xyzIn = nx();
  const float* tW0 = nx(); const float* tW1 = nx(); const float* tW2 = nx();
  const float* tb0 = nx(); const float* tb1 = nx(); const float* tb2 = nx();

  struct CicP { const float *W1,*W2,*Wa,*Ws,*Wsc,*b1,*b2,*bsc,*g1,*g2,*gsc,
                            *lWm,*lWx,*lbm,*lbx,*lgm,*lgx; } cp[8];
  struct Cfg { int np; float rad; int ic, oc, pl; int curve; };
  const Cfg cfgs[8] = {
    {1024, 0.05f,  32,  64,  16, 1}, {1024, 0.05f,  64,  64,  16, 1},
    {1024, 0.05f,  64, 128,  32, 1}, {1024, 0.10f, 128, 128,  32, 1},
    { 256, 0.10f, 128, 256, 128, 0}, { 256, 0.20f, 256, 256,  64, 0},
    {  64, 0.20f, 256, 512, 256, 0}, {  64, 0.40f, 512, 512, 128, 0}};
  for (int i = 0; i < 8; ++i) {
    bool cur = cfgs[i].curve != 0;
    bool sc  = cfgs[i].ic != cfgs[i].oc;
    CicP& p = cp[i];
    p.W1 = nx(); p.W2 = nx();
    if (cur) { p.Wa = nx(); p.Ws = nx(); } else { p.Wa = nullptr; p.Ws = nullptr; }
    p.Wsc = sc ? nx() : nullptr;
    p.b1 = nx(); p.b2 = nx();
    p.bsc = sc ? nx() : nullptr;
    p.g1 = nx(); p.g2 = nx();
    p.gsc = sc ? nx() : nullptr;
    p.lWm = nx(); p.lWx = nx(); p.lbm = nx(); p.lbx = nx(); p.lgm = nx(); p.lgx = nx();
  }
  const float* tg0 = nx(); const float* tg1 = nx();
  const float* l0W = nx(); const float* l0b = nx(); const float* l0g = nx();

  // ---- workspace carve-up ----
  char* base = (char*)d_ws;
  size_t off = 0;
  auto alloc = [&](size_t bytes) {
    void* p = base + off;
    off += (bytes + 255) & ~(size_t)255;
    return p;
  };
  float* xyzA   = (float*)alloc((size_t)16 * 1024 * 3 * 4);
  float* xyzB   = (float*)alloc((size_t)16 * 1024 * 3 * 4);
  float* xbufA  = (float*)alloc((size_t)2097152 * 4);
  float* xbufB  = (float*)alloc((size_t)2097152 * 4);
  float* bufH   = (float*)alloc((size_t)2097152 * 4);
  float* bufG   = (float*)alloc((size_t)2097152 * 4);
  float* bufH2  = (float*)alloc((size_t)2097152 * 4);
  float* f9b    = (float*)alloc((size_t)2949120 * 4);   // [B*1024*20, 9]
  float* hk1    = (float*)alloc((size_t)10485760 * 4);  // [B*N*20, pl] staging
  float* sco    = (float*)alloc((size_t)32768 * 4);
  float* cmean  = (float*)alloc((size_t)51200 * 4);
  float* fpool  = (float*)alloc((size_t)32768 * 4);
  float* f2b    = (float*)alloc((size_t)8192 * 4);
  int*   idxb   = (int*)alloc((size_t)655360 * 4);      // [B*2048*20]
  int*   nbrrow = (int*)alloc((size_t)327680 * 4);
  int*   sidx   = (int*)alloc((size_t)16384 * 4);
  int*   srow   = (int*)alloc((size_t)16384 * 4);
  int*   starts = (int*)alloc((size_t)1600 * 4);

  // ---- stage 0: lpfa_initial on the raw cloud ----
  int Ncur = 2048;
  knn_kernel<<<g1d(16 * Ncur), 256, 0, stream>>>(xyzIn, xyzIn, Ncur, Ncur, idxb);
  lpfa0_kernel<<<g1d(16 * Ncur * 32), 256, 0, stream>>>(xyzIn, idxb, l0W, l0g, l0b,
                                                        Ncur, xbufA);
  const float* xyzCur = xyzIn;
  float* xCur = xbufA;
  float* xAlt = xbufB;
  float* xyzSlots[2] = {xyzA, xyzB};
  int xyzSlot = 0;

  // ---- 8 CIC blocks ----
  for (int i = 0; i < 8; ++i) {
    const Cfg c = cfgs[i];
    const CicP& P = cp[i];

    if (c.np < Ncur) {  // masked max pool (FPS + kNN + radius mask)
      fps_kernel<<<16, 256, 0, stream>>>(xyzCur, Ncur, c.np, sidx);
      build_srow_kernel<<<g1d(16 * c.np), 256, 0, stream>>>(sidx, c.np, Ncur, srow);
      float* nxyz = xyzSlots[xyzSlot]; xyzSlot ^= 1;
      gather_rows_kernel<<<g1d(16 * c.np * 3), 256, 0, stream>>>(xyzCur, srow,
                                                                 16 * c.np, 3, nxyz);
      knn_kernel<<<g1d(16 * c.np), 256, 0, stream>>>(nxyz, xyzCur, c.np, Ncur, idxb);
      masked_pool_kernel<<<g1d(16 * c.np * c.ic), 256, 0, stream>>>(
          nxyz, xyzCur, idxb, xCur, c.np, Ncur, c.ic, c.rad * c.rad, xAlt);
      xyzCur = nxyz; Ncur = c.np;
      float* t = xCur; xCur = xAlt; xAlt = t;
    }

    // self-kNN for this level (curve walk + lpfa share it)
    knn_kernel<<<g1d(16 * Ncur), 256, 0, stream>>>(xyzCur, xyzCur, Ncur, Ncur, idxb);

    const int rows = 16 * Ncur;
    // h = relu(bn(x @ W1))  [WMMA]
    gemmL(stream, xCur, P.W1, bufH, rows, c.pl, c.ic, P.g1, P.b1, nullptr, nullptr, 1, 1);
    float* hptr = bufH;

    if (c.curve) {  // CurveAggregation
      score_kernel<<<g1d(rows), 256, 0, stream>>>(bufH, P.Ws, rows, c.pl, sco);
      topk_starts_kernel<<<16, 256, 0, stream>>>(sco, Ncur, 100, starts);
      curve_walk_kernel<<<g1d(16 * 100), 256, 0, stream>>>(bufH, idxb, starts,
                                                           Ncur, c.pl, 100, 5, cmean);
      curve_att_kernel<<<g1d(rows), 256, 0, stream>>>(bufH, cmean, Ncur, c.pl, 100, bufG);
      // h = h + agg @ Wa  [WMMA, fused residual]
      gemmL(stream, bufG, P.Wa, bufH2, rows, c.pl, c.pl, nullptr, nullptr,
            bufH, nullptr, 0, 1);
      hptr = bufH2;
    }

    // lpfa: pe = bn(f9@Wx); h1 = lrelu(gather(h, idx) + pe)  [WMMA, fused gather]
    relpos_kernel<<<g1d(rows * KNN), 256, 0, stream>>>(xyzCur, idxb, Ncur, f9b, nbrrow);
    const int rk = rows * KNN;
    gemmL(stream, f9b, P.lWx, hk1, rk, c.pl, 9, P.lgx, P.lbx, hptr, nbrrow, 2, 1);
    // h2 = lrelu(bn(h1@Wm)); mean over k fused via atomic epilogue  [WMMA]
    zero_kernel<<<g1d(rows * c.pl), 256, 0, stream>>>(bufG, rows * c.pl);
    gemmL(stream, hk1, P.lWm, bufG, rk, c.pl, c.pl, P.lgm, P.lbm, nullptr, nullptr, 2, KNN);

    // out = relu(bn(h@W2) + shortcut)  [WMMA, fused residual + relu]
    const float* scp = xCur;
    if (c.ic != c.oc) {
      gemmL(stream, xCur, P.Wsc, bufH2, rows, c.oc, c.ic, P.gsc, P.bsc,
            nullptr, nullptr, 0, 1);
      scp = bufH2;
    }
    gemmL(stream, bufG, P.W2, xAlt, rows, c.oc, c.pl, P.g2, P.b2, scp, nullptr, 1, 1);
    float* t = xCur; xCur = xAlt; xAlt = t;
  }

  // ---- head: x[16,64,512] -> logits[16,40] ----
  gemmL(stream, xCur, tW0, bufH, 16 * 64, 1024, 512, tg0, tb0, nullptr, nullptr, 1, 1);
  final_pool_kernel<<<g1d(16 * 1024), 256, 0, stream>>>(bufH, 1024, fpool);
  gemmL(stream, fpool, tW1, f2b, 16, 512, 2048, tg1, tb1, nullptr, nullptr, 1, 1);
  gemmL(stream, f2b, tW2, (float*)d_out, 16, 40, 512, nullptr, tb2, nullptr, nullptr, 0, 1);
}
